// HybridTwoTower_65506841198662
// MI455X (gfx1250) — matrix operand order repaired and checked by reading, not compiled, not run on hardware
//
#include <hip/hip_runtime.h>

// ---------------------------------------------------------------------------
// MI455X (gfx1250) two-tower recsys forward.
//  * bf16 WMMA (v_wmma_f32_16x16x32_bf16) for the item MLP and user MLP.
//  * attention rewritten as scores[h] = V[h]·(Wk q) + q·bk  (exact), so the
//    51 GFLOP K-projection collapses to a per-user 256x128 GEMV; the bf16
//    item-embedding table (51 MB) stays resident in the 192 MB L2.
//  * per-user gather uses GLOBAL_LOAD_ASYNC_TO_LDS_B128 (ASYNCcnt), issued
//    before the q / p GEMVs so the DMA overlaps compute.
//  * wave32 tiling: one wave owns a 16-row x 256-col output tile;
//    B fragments come from pre-transposed bf16 weights (two 16B loads/lane).
// ---------------------------------------------------------------------------

typedef __attribute__((ext_vector_type(16))) __bf16 v16bf;
typedef __attribute__((ext_vector_type(8)))  __bf16 v8bf;
typedef __attribute__((ext_vector_type(8)))  float  v8f;
typedef __attribute__((ext_vector_type(4)))  float  v4f;

#define N_ITEMS 100000
#define N_USERS 16384
#define HLEN    48
#define GCN_D   128
#define ATTN_D  128
#define HID_D   256
#define TOW_D   256

__device__ __forceinline__ v8f wmma_bf16(v16bf a, v16bf b, v8f c) {
  // (neg_a, A, neg_b, B, c_mod, C, reuse_a, reuse_b)
  return __builtin_amdgcn_wmma_f32_16x16x32_bf16(false, a, false, b,
                                                 (short)0, c, false, false);
}

// A fragment (16x32 bf16): lane holds row (lane&15); elements 0..7 at
// row[kb+koff+e], 8..15 at row[kb+16+koff+e], koff = 8*(lane>=16).
__device__ __forceinline__ v16bf load_fragA_bf16(const __bf16* rowk, int koff) {
  v8bf lo = *reinterpret_cast<const v8bf*>(rowk + koff);
  v8bf hi = *reinterpret_cast<const v8bf*>(rowk + 16 + koff);
  v16bf r;
#pragma unroll
  for (int e = 0; e < 8; ++e) { r[e] = lo[e]; r[e + 8] = hi[e]; }
  return r;
}

// Same fragment but sourced from f32 data with on-the-fly bf16 convert.
__device__ __forceinline__ v16bf load_fragA_f32(const float* rowk, int koff) {
  v4f a0 = *reinterpret_cast<const v4f*>(rowk + koff);
  v4f a1 = *reinterpret_cast<const v4f*>(rowk + koff + 4);
  v4f b0 = *reinterpret_cast<const v4f*>(rowk + 16 + koff);
  v4f b1 = *reinterpret_cast<const v4f*>(rowk + 16 + koff + 4);
  v16bf r;
#pragma unroll
  for (int e = 0; e < 4; ++e) {
    r[e]      = (__bf16)a0[e];
    r[e + 4]  = (__bf16)a1[e];
    r[e + 8]  = (__bf16)b0[e];
    r[e + 12] = (__bf16)b1[e];
  }
  return r;
}

// B fragment (32x16 bf16): lane holds column (lane&15); elements 0..7 at
// col[kb+koff+e], 8..15 at col[kb+16+koff+e], koff = 8*(lane>=16).
__device__ __forceinline__ v16bf load_fragB(const __bf16* colk, int kboff) {
  const int koff = kboff >> 1;
  v8bf lo = *reinterpret_cast<const v8bf*>(colk + koff);
  v8bf hi = *reinterpret_cast<const v8bf*>(colk + 16 + koff);
  v16bf r;
#pragma unroll
  for (int e = 0; e < 8; ++e) { r[e] = lo[e]; r[e + 8] = hi[e]; }
  return r;
}

// ---------------------------------------------------------------------------
// Kernel P: transpose weights to bf16 column-major so B fragments are
// contiguous. 262144 elements, grid 1024 x 256.
// ---------------------------------------------------------------------------
__global__ __launch_bounds__(256)
void prep_weights_kernel(const float* __restrict__ Wi1, const float* __restrict__ Wi2,
                         const float* __restrict__ Wu1, const float* __restrict__ Wu2,
                         __bf16* __restrict__ Wi1g_t, __bf16* __restrict__ Wi2_t,
                         __bf16* __restrict__ Wu1_t,  __bf16* __restrict__ Wu2_t) {
  int i = blockIdx.x * 256 + threadIdx.x;
  if (i < 256 * 128) {               // Wi1g_t[n][k] = Wi1[k+2][n]  (gcn rows)
    int n = i >> 7, k = i & 127;
    Wi1g_t[i] = (__bf16)Wi1[(k + 2) * HID_D + n];
    return;
  }
  i -= 256 * 128;
  if (i < 256 * 256) {               // Wi2_t[n][k] = Wi2[k][n]
    int n = i >> 8, k = i & 255;
    Wi2_t[i] = (__bf16)Wi2[k * TOW_D + n];
    return;
  }
  i -= 256 * 256;
  if (i < 256 * 384) {               // Wu1_t[n][k] = Wu1[k][n], k<128 gcn, rest hist
    int n = i / 384, k = i % 384;
    Wu1_t[i] = (__bf16)Wu1[k * HID_D + n];
    return;
  }
  i -= 256 * 384;
  {                                  // Wu2_t[n][k] = Wu2[k][n]
    int n = i >> 8, k = i & 255;
    Wu2_t[i] = (__bf16)Wu2[k * TOW_D + n];
  }
}

// ---------------------------------------------------------------------------
// Kernel I: item encoder. One wave per 16 items (6250 blocks x 32 threads).
// h = relu([feat, gcn] @ Wi1 + bi1); item_emb = l2norm(h @ Wi2 + bi2) -> bf16.
// The K=2 feat contribution is folded into the accumulator init.
// ---------------------------------------------------------------------------
__global__ __launch_bounds__(32)
void item_encode_kernel(const float* __restrict__ item_feat,
                        const float* __restrict__ gcn_item,
                        const float* __restrict__ Wi1,
                        const float* __restrict__ bi1,
                        const float* __restrict__ bi2,
                        const __bf16* __restrict__ Wi1g_t,
                        const __bf16* __restrict__ Wi2_t,
                        __bf16* __restrict__ item_emb) {
  __shared__ alignas(16) __bf16 h1[16 * HID_D];   // layer-1 activations (bf16)
  __shared__ alignas(16) float  y2[16 * TOW_D];   // layer-2 pre-norm (f32)
  __shared__ float  sc[16];                       // per-row 1/||y||

  const int lane  = threadIdx.x;
  const int half  = lane >> 4;
  const int lm    = lane & 15;
  const int koff  = half * 8;
  const int kboff = half * 16;
  const int m0    = blockIdx.x * 16;

  // Layer-1 A fragments: gcn row (K=128 -> 4 K-steps), f32 -> bf16 on the fly.
  const float* arow = gcn_item + (size_t)(m0 + lm) * GCN_D;
  v16bf a1[4];
#pragma unroll
  for (int ks = 0; ks < 4; ++ks) a1[ks] = load_fragA_f32(arow + ks * 32, koff);

  // item_feat values for the 8 C/D rows this lane owns.
  float f0r[8], f1r[8];
#pragma unroll
  for (int r = 0; r < 8; ++r) {
    const int m = m0 + half * 8 + r;
    f0r[r] = item_feat[2 * m + 0];
    f1r[r] = item_feat[2 * m + 1];
  }

  for (int nt = 0; nt < 16; ++nt) {
    const int col = nt * 16 + lm;
    const float w0 = Wi1[0 * HID_D + col];
    const float w1 = Wi1[1 * HID_D + col];
    const float bb = bi1[col];
    v8f c;
#pragma unroll
    for (int r = 0; r < 8; ++r) c[r] = bb + f0r[r] * w0 + f1r[r] * w1;
    const __bf16* bcol = Wi1g_t + (size_t)col * GCN_D;
#pragma unroll
    for (int ks = 0; ks < 4; ++ks)
      c = wmma_bf16(a1[ks], load_fragB(bcol + ks * 32, kboff), c);
#pragma unroll
    for (int r = 0; r < 8; ++r)
      h1[(half * 8 + r) * HID_D + col] = (__bf16)fmaxf(c[r], 0.0f);
  }

  asm volatile("s_wait_dscnt 0" ::: "memory");   // LDS writes visible to wave

  // Layer 2: K=256 -> 8 K-steps, A from LDS.
  v16bf a2[8];
  const __bf16* hrow = &h1[lm * HID_D];
#pragma unroll
  for (int ks = 0; ks < 8; ++ks) a2[ks] = load_fragA_bf16(hrow + ks * 32, koff);

  float ss[8] = {0.f, 0.f, 0.f, 0.f, 0.f, 0.f, 0.f, 0.f};
  for (int nt = 0; nt < 16; ++nt) {
    const int col = nt * 16 + lm;
    const float bb = bi2[col];
    v8f c;
#pragma unroll
    for (int r = 0; r < 8; ++r) c[r] = bb;
    const __bf16* bcol = Wi2_t + (size_t)col * HID_D;
#pragma unroll
    for (int ks = 0; ks < 8; ++ks)
      c = wmma_bf16(a2[ks], load_fragB(bcol + ks * 32, kboff), c);
#pragma unroll
    for (int r = 0; r < 8; ++r) {
      ss[r] += c[r] * c[r];
      y2[(half * 8 + r) * TOW_D + col] = c[r];
    }
  }

  // Row L2 norms: lanes of a row differ only in bits [3:0].
#pragma unroll
  for (int r = 0; r < 8; ++r) {
    float s = ss[r];
    s += __shfl_xor(s, 1);
    s += __shfl_xor(s, 2);
    s += __shfl_xor(s, 4);
    s += __shfl_xor(s, 8);
    sc[half * 8 + r] = 1.0f / fmaxf(sqrtf(s), 1e-12f);
  }

  asm volatile("s_wait_dscnt 0" ::: "memory");

  // Normalize + store, 16B (8 x bf16) per lane-iteration.
  for (int j = lane; j < (16 * TOW_D) / 8; j += 32) {
    const int lr = j >> 5;                 // (j*8) / 256
    const float* ysrc = &y2[j * 8];
    const float s = sc[lr];
    v8bf ov;
#pragma unroll
    for (int e = 0; e < 8; ++e) ov[e] = (__bf16)(ysrc[e] * s);
    *reinterpret_cast<v8bf*>(item_emb + (size_t)m0 * TOW_D + j * 8) = ov;
  }
}

// ---------------------------------------------------------------------------
// Kernel A: per-user attention via projected query. One 128-thread block per
// user. scores[h] = V[h]·p + q·bk with p = Wk q.  V is gathered from the
// L2-resident bf16 item table straight into LDS with
// GLOBAL_LOAD_ASYNC_TO_LDS_B128, overlapped with the q and p GEMVs.
// ---------------------------------------------------------------------------
__global__ __launch_bounds__(128)
void user_attn_kernel(const float* __restrict__ gcn_user,
                      const int* __restrict__ hist_idx,
                      const int* __restrict__ hist_len,
                      const float* __restrict__ Wq,
                      const float* __restrict__ bq,
                      const float* __restrict__ Wk,
                      const float* __restrict__ bk,
                      const __bf16* __restrict__ item_emb,
                      float* __restrict__ hist_out) {
  __shared__ float  u_s[GCN_D];
  __shared__ float  q_s[ATTN_D];
  __shared__ float  p_s[TOW_D];
  __shared__ alignas(16) __bf16 V[HLEN * TOW_D];
  __shared__ int    hidx[HLEN];
  __shared__ float  sc_s[HLEN];
  __shared__ float  aw[HLEN];

  const int u = blockIdx.x;
  const int tid = threadIdx.x;

  u_s[tid] = gcn_user[(size_t)u * GCN_D + tid];
  if (tid < HLEN) hidx[tid] = hist_idx[u * HLEN + tid];
  __syncthreads();

  // --- Kick off the async gather: 48 rows x 512B, 16B chunks, 12/thread.
  // LDS target address = low 32 bits of the generic shared pointer
  // (flat LDS aperture truncation, ISA 10.2).
  {
    const unsigned lds_base = (unsigned)(unsigned long long)(&V[0]);
    for (int j = tid; j < HLEN * (TOW_D / 8); j += 128) {
      const int h = j >> 5, ck = j & 31;
      const unsigned long long g =
          (unsigned long long)(item_emb + (size_t)hidx[h] * TOW_D + ck * 8);
      const unsigned l = lds_base + (unsigned)(j * 16);
      asm volatile("global_load_async_to_lds_b128 %0, %1, off"
                   :: "v"(l), "v"(g) : "memory");
    }
  }

  // --- Overlapped compute: q = u @ Wq + bq  (thread tid owns q[tid]).
  float q = bq[tid];
  for (int d = 0; d < GCN_D; ++d) q = fmaf(u_s[d], Wq[d * ATTN_D + tid], q);
  q_s[tid] = q;
  __syncthreads();

  // p[t] = sum_a Wk[t][a] q[a];  c0 = bk · q
  float p0 = 0.f, p1 = 0.f, c0 = 0.f;
  const float* wk0 = Wk + (size_t)tid * ATTN_D;
  const float* wk1 = Wk + (size_t)(tid + 128) * ATTN_D;
  for (int a = 0; a < ATTN_D; ++a) {
    const float qa = q_s[a];
    p0 = fmaf(wk0[a], qa, p0);
    p1 = fmaf(wk1[a], qa, p1);
    c0 = fmaf(bk[a], qa, c0);
  }
  p_s[tid] = p0;
  p_s[tid + 128] = p1;

  // Drain this wave's async copies, then block-wide barrier before reading V.
  asm volatile("s_wait_asynccnt 0x0" ::: "memory");
  __syncthreads();

  const int hl = hist_len[u];
  if (tid < HLEN) {
    float s = c0;
    const __bf16* vr = &V[tid * TOW_D];
    for (int t = 0; t < TOW_D; ++t) s = fmaf((float)vr[t], p_s[t], s);
    sc_s[tid] = (tid <= hl) ? s * 0.088388347648318447f /* 1/sqrt(128) */
                            : -1e30f;
  }
  __syncthreads();

  float mx = -1e30f;
  for (int h = 0; h < HLEN; ++h) mx = fmaxf(mx, sc_s[h]);
  float sum = 0.f;
  for (int h = 0; h < HLEN; ++h) sum += __expf(sc_s[h] - mx);
  if (tid < HLEN) aw[tid] = __expf(sc_s[tid] - mx) / sum;
  __syncthreads();

  // hist[t] = sum_h aw[h] * V[h][t]
#pragma unroll
  for (int rep = 0; rep < 2; ++rep) {
    const int t = tid + rep * 128;
    float acc = 0.f;
    for (int h = 0; h < HLEN; ++h)
      acc = fmaf(aw[h], (float)V[h * TOW_D + t], acc);
    hist_out[(size_t)u * TOW_D + t] = acc;
  }
}

// ---------------------------------------------------------------------------
// Kernel U: user MLP. One wave per 16 users (1024 blocks x 32). K=384 layer 1
// reads the [gcn | hist] concat from two sources (4 + 8 K-steps).
// ---------------------------------------------------------------------------
__global__ __launch_bounds__(32)
void user_mlp_kernel(const float* __restrict__ gcn_user,
                     const float* __restrict__ hist,
                     const float* __restrict__ bu1,
                     const float* __restrict__ bu2,
                     const __bf16* __restrict__ Wu1_t,
                     const __bf16* __restrict__ Wu2_t,
                     float* __restrict__ out) {
  __shared__ alignas(16) __bf16 h1[16 * HID_D];
  __shared__ alignas(16) float  y2[16 * TOW_D];
  __shared__ float  sc[16];

  const int lane  = threadIdx.x;
  const int half  = lane >> 4;
  const int lm    = lane & 15;
  const int koff  = half * 8;
  const int kboff = half * 16;
  const int m0    = blockIdx.x * 16;

  v16bf a1[12];
  const float* ur = gcn_user + (size_t)(m0 + lm) * GCN_D;
  const float* hr = hist + (size_t)(m0 + lm) * TOW_D;
#pragma unroll
  for (int ks = 0; ks < 4; ++ks) a1[ks] = load_fragA_f32(ur + ks * 32, koff);
#pragma unroll
  for (int ks = 0; ks < 8; ++ks) a1[4 + ks] = load_fragA_f32(hr + ks * 32, koff);

  for (int nt = 0; nt < 16; ++nt) {
    const int col = nt * 16 + lm;
    const float bb = bu1[col];
    v8f c;
#pragma unroll
    for (int r = 0; r < 8; ++r) c[r] = bb;
    const __bf16* bcol = Wu1_t + (size_t)col * 384;
#pragma unroll
    for (int ks = 0; ks < 12; ++ks)
      c = wmma_bf16(a1[ks], load_fragB(bcol + ks * 32, kboff), c);
#pragma unroll
    for (int r = 0; r < 8; ++r)
      h1[(half * 8 + r) * HID_D + col] = (__bf16)fmaxf(c[r], 0.0f);
  }

  asm volatile("s_wait_dscnt 0" ::: "memory");

  v16bf a2[8];
  const __bf16* hrow = &h1[lm * HID_D];
#pragma unroll
  for (int ks = 0; ks < 8; ++ks) a2[ks] = load_fragA_bf16(hrow + ks * 32, koff);

  float ss[8] = {0.f, 0.f, 0.f, 0.f, 0.f, 0.f, 0.f, 0.f};
  for (int nt = 0; nt < 16; ++nt) {
    const int col = nt * 16 + lm;
    const float bb = bu2[col];
    v8f c;
#pragma unroll
    for (int r = 0; r < 8; ++r) c[r] = bb;
    const __bf16* bcol = Wu2_t + (size_t)col * HID_D;
#pragma unroll
    for (int ks = 0; ks < 8; ++ks)
      c = wmma_bf16(a2[ks], load_fragB(bcol + ks * 32, kboff), c);
#pragma unroll
    for (int r = 0; r < 8; ++r) {
      ss[r] += c[r] * c[r];
      y2[(half * 8 + r) * TOW_D + col] = c[r];
    }
  }

#pragma unroll
  for (int r = 0; r < 8; ++r) {
    float s = ss[r];
    s += __shfl_xor(s, 1);
    s += __shfl_xor(s, 2);
    s += __shfl_xor(s, 4);
    s += __shfl_xor(s, 8);
    sc[half * 8 + r] = 1.0f / fmaxf(sqrtf(s), 1e-12f);
  }

  asm volatile("s_wait_dscnt 0" ::: "memory");

  // Normalize + store, 16B (4 x f32) per lane-iteration.
  for (int j = lane; j < (16 * TOW_D) / 4; j += 32) {
    const int lr = j >> 6;                 // (j*4) / 256
    v4f yv = *reinterpret_cast<const v4f*>(&y2[j * 4]);
    const float s = sc[lr];
    v4f ov;
#pragma unroll
    for (int e = 0; e < 4; ++e) ov[e] = yv[e] * s;
    *reinterpret_cast<v4f*>(out + (size_t)m0 * TOW_D + j * 4) = ov;
  }
}

// ---------------------------------------------------------------------------
// Host launcher
// ---------------------------------------------------------------------------
extern "C" void kernel_launch(void* const* d_in, const int* in_sizes, int n_in,
                              void* d_out, int out_size, void* d_ws, size_t ws_size,
                              hipStream_t stream) {
  (void)in_sizes; (void)n_in; (void)out_size; (void)ws_size;

  const float* item_feat = (const float*)d_in[0];
  const float* gcn_item  = (const float*)d_in[1];
  const float* gcn_user  = (const float*)d_in[2];
  const int*   hist_idx  = (const int*)d_in[3];
  const int*   hist_len  = (const int*)d_in[4];
  const float* Wi1 = (const float*)d_in[5];
  const float* bi1 = (const float*)d_in[6];
  const float* Wi2 = (const float*)d_in[7];
  const float* bi2 = (const float*)d_in[8];
  const float* Wq  = (const float*)d_in[9];
  const float* bq  = (const float*)d_in[10];
  const float* Wk  = (const float*)d_in[11];
  const float* bk  = (const float*)d_in[12];
  const float* Wu1 = (const float*)d_in[13];
  const float* bu1 = (const float*)d_in[14];
  const float* Wu2 = (const float*)d_in[15];
  const float* bu2 = (const float*)d_in[16];
  float* out = (float*)d_out;

  char* ws = (char*)d_ws;
  size_t off = 0;
  auto wsget = [&](size_t bytes) -> void* {
    void* p = (void*)(ws + off);
    off += (bytes + 255) & ~((size_t)255);
    return p;
  };
  __bf16* item_emb_bf = (__bf16*)wsget((size_t)N_ITEMS * TOW_D * 2);  // 51.2 MB
  float*  hist_buf    = (float*) wsget((size_t)N_USERS * TOW_D * 4);  // 16.8 MB
  __bf16* Wi1g_t = (__bf16*)wsget((size_t)256 * 128 * 2);
  __bf16* Wi2_t  = (__bf16*)wsget((size_t)256 * 256 * 2);
  __bf16* Wu1_t  = (__bf16*)wsget((size_t)256 * 384 * 2);
  __bf16* Wu2_t  = (__bf16*)wsget((size_t)256 * 256 * 2);

  prep_weights_kernel<<<1024, 256, 0, stream>>>(Wi1, Wi2, Wu1, Wu2,
                                                Wi1g_t, Wi2_t, Wu1_t, Wu2_t);
  item_encode_kernel<<<N_ITEMS / 16, 32, 0, stream>>>(
      item_feat, gcn_item, Wi1, bi1, bi2, Wi1g_t, Wi2_t, item_emb_bf);
  user_attn_kernel<<<N_USERS, 128, 0, stream>>>(
      gcn_user, hist_idx, hist_len, Wq, bq, Wk, bk, item_emb_bf, hist_buf);
  user_mlp_kernel<<<N_USERS / 16, 32, 0, stream>>>(
      gcn_user, hist_buf, bu1, bu2, Wu1_t, Wu2_t, out);
}
